// HybridNATModel_65481071408093
// MI455X (gfx1250) — matrix operand order repaired and checked by reading, not compiled, not run on hardware
//
#include <hip/hip_runtime.h>
#include <hip/hip_bf16.h>

// ---------------------------------------------------------------------------
// HybridNAT forward for MI455X (gfx1250, wave32).
// conv2 and fc1 run on v_wmma_f32_16x16x32_f16 (f32 accumulation).
// All reductions are fixed-order (no float atomics) => deterministic.
// Round 2: removed all exec-masked per-element loads in the WMMA feed paths
// (LDS zero-region for invalid im2col K, pre-padded conv2 weights).
// ---------------------------------------------------------------------------

typedef _Float16 h8  __attribute__((ext_vector_type(8)));
typedef _Float16 v16h __attribute__((ext_vector_type(16)));
typedef float    v8f  __attribute__((ext_vector_type(8)));

union HU { v16h v; h8 h[2]; };

#define BATCH 16384

// ---------------------------------------------------------------------------
// K0: pre-swizzle fc1_w [64,787] f32 into WMMA B-operand layout, f16,
// K padded to 800.  Layout: w1h[ ((s*2+hi)*64 + n)*16 + e ],  k = (s*2+hi)*16+e
// so each lane's 16 B elements are 32 contiguous bytes (two b128 loads).
// ---------------------------------------------------------------------------
__global__ void k0_prep_fc1(const float* __restrict__ fc1_w,
                            _Float16* __restrict__ w1h) {
  int i = blockIdx.x * 256 + threadIdx.x;            // 800*64 = 51200 entries
  if (i >= 800 * 64) return;
  int e  = i & 15;
  int n  = (i >> 4) & 63;
  int sh = i >> 10;                                  // 0..49
  int k  = sh * 16 + e;                              // 0..799
  float v = (k < 787) ? fc1_w[n * 787 + k] : 0.0f;
  w1h[i] = (_Float16)v;
}

// ---------------------------------------------------------------------------
// K0b: zero-pad conv2 weights [16][72] f32 -> w2p[16][96] f16 so the WMMA A
// operand is loadable with unconditional 128-bit loads.
// ---------------------------------------------------------------------------
__global__ void k0b_prep_conv2(const float* __restrict__ w2,
                               _Float16* __restrict__ w2p) {
  int i = blockIdx.x * 256 + threadIdx.x;            // 16*96 = 1536 entries
  if (i >= 16 * 96) return;
  int o = i / 96, k = i - 96 * o;
  float v = (k < 72) ? w2[o * 72 + k] : 0.0f;
  w2p[i] = (_Float16)v;
}

// ---------------------------------------------------------------------------
// K1: conv1(3x3,pad1) + relu + maxpool2 -> f1p[B][8][16][16] f16 with a
// built-in zero ring (padded 14x14 plane) so K2's im2col gather is branchless.
// One block per image; x plane staged zero-padded (30x30) in LDS.
// ---------------------------------------------------------------------------
__global__ void __launch_bounds__(256)
k1_conv1(const float* __restrict__ x, const float* __restrict__ w1,
         const float* __restrict__ b1, _Float16* __restrict__ f1p) {
  __shared__ float xs[30 * 30];
  int b = blockIdx.x;
  int tid = threadIdx.x;
  const float* xb = x + (size_t)b * 784;
  for (int i = tid; i < 900; i += 256) {
    int r = i / 30, c = i % 30;
    float v = 0.0f;
    if (r >= 1 && r <= 28 && c >= 1 && c <= 28) v = xb[(r - 1) * 28 + (c - 1)];
    xs[i] = v;
  }
  __syncthreads();

  int r = tid >> 4, cc = tid & 15;                   // coords in padded 16x16
  bool border = (r == 0) || (r == 15) || (cc == 0) || (cc == 15);
  int ph = r - 1, pw = cc - 1;                       // pooled coords 0..13
  _Float16* outb = f1p + (size_t)b * 2048 + tid;
  for (int c = 0; c < 8; ++c) {
    float m = 0.0f;
    if (!border) {
      float wr[9];
#pragma unroll
      for (int t = 0; t < 9; ++t) wr[t] = w1[c * 9 + t];
      float bias = b1[c];
      m = -3.4e38f;
#pragma unroll
      for (int dy = 0; dy < 2; ++dy)
#pragma unroll
        for (int dx = 0; dx < 2; ++dx) {
          int oh = 2 * ph + dy, ow = 2 * pw + dx;    // conv-out coords 0..27
          float acc = bias;
#pragma unroll
          for (int i = 0; i < 3; ++i)
#pragma unroll
            for (int j = 0; j < 3; ++j)
              acc += wr[i * 3 + j] * xs[(oh + i) * 30 + (ow + j)];
          m = fmaxf(m, acc);
        }
      m = fmaxf(m, 0.0f);                            // max(relu) == relu(max)
    }
    outb[c * 256] = (_Float16)m;
  }
}

// ---------------------------------------------------------------------------
// K2: conv2 as implicit GEMM on WMMA.  Per image:
//   out[16 oc][196 px] = W[16][K=72->96] x Im2col[96][196]
// 13 pixel tiles x 3 K-steps of v_wmma_f32_16x16x32_f16, bias+relu in
// C-layout, then 2x2 maxpool via LDS -> combined[b][0..783] f16.
// Block = 128 threads (4 waves), one image per block.  The staged image has a
// 512-entry zero region appended; invalid im2col K elements gather from it so
// every B-operand element is one unconditional ds_load_u16 (no exec masking).
// ---------------------------------------------------------------------------
__global__ void __launch_bounds__(128)
k2_conv2(const _Float16* __restrict__ f1p, const _Float16* __restrict__ w2p,
         const float* __restrict__ b2, _Float16* __restrict__ combined) {
  __shared__ _Float16 f1s[2560];                     // 2048 data + 512 zeros
  __shared__ float ys[16 * 208];                     // pre-pool relu output
  int b = blockIdx.x;
  int tid = threadIdx.x;
  int lane = tid & 31, wv = tid >> 5;
  int l15 = lane & 15, hi = lane >> 4;

  { // stage the image (4 KB) into LDS + zero-fill the overflow region
    const unsigned* src = (const unsigned*)(f1p + (size_t)b * 2048);
    unsigned* dst = (unsigned*)f1s;
    for (int i = tid; i < 1280; i += 128) dst[i] = (i < 1024) ? src[i] : 0u;
  }
  __syncthreads();

  // A operand (weights): M = out-channel = l15, K(e) = (e>>3)*16 + hi*8 + (e&7)
  // w2p is zero-padded to K=96 -> two unconditional 16B loads per K-step.
  v16h A[3];
#pragma unroll
  for (int s = 0; s < 3; ++s) {
    HU a;
    const h8* wp = (const h8*)(w2p + l15 * 96 + s * 32 + hi * 8);
    a.h[0] = wp[0];                                  // k = s*32+hi*8    .. +7
    a.h[1] = wp[2];                                  // k = s*32+16+hi*8 .. +7
    A[s] = a.v;
  }
  // B gather offsets: K(e) = (e>>3)*16 + hi*8 + (e&7) ;  addr = c*256 + dh*16 + dw + pbase
  // invalid K (>=72) -> 2176 (zero region; 2176 + max pbase 221 < 2560)
  int boff[3][16];
#pragma unroll
  for (int s = 0; s < 3; ++s)
#pragma unroll
    for (int e = 0; e < 16; ++e) {
      int k = s * 32 + (e >> 3) * 16 + hi * 8 + (e & 7);
      int c = k / 9, r = k - 9 * c, dh = r / 3, dw = r - 3 * dh;
      boff[s][e] = (k < 72) ? (c * 256 + dh * 16 + dw) : 2176;
    }
  float b2s[8];
#pragma unroll
  for (int v = 0; v < 8; ++v) b2s[v] = b2[(hi << 3) + v];

  for (int t = wv; t < 13; t += 4) {                 // uniform per wave
    int pn = t * 16 + l15;                           // GEMM column = pixel
    int p = pn > 195 ? 195 : pn;                     // clamp tail lanes
    int ph = p / 14, pw = p - 14 * ph;
    int pbase = ph * 16 + pw;
    v8f C = {0.f, 0.f, 0.f, 0.f, 0.f, 0.f, 0.f, 0.f};
#pragma unroll
    for (int s = 0; s < 3; ++s) {
      v16h Bv;
#pragma unroll
      for (int e = 0; e < 16; ++e) Bv[e] = f1s[boff[s][e] + pbase];
      C = __builtin_amdgcn_wmma_f32_16x16x32_f16(false, A[s], false, Bv,
                                                 (short)0, C, false, false);
    }
#pragma unroll
    for (int v = 0; v < 8; ++v) {                    // D: M = v + 8*hi, N = l15
      float val = C[v] + b2s[v];
      ys[((hi << 3) + v) * 208 + pn] = fmaxf(val, 0.0f);
    }
  }
  __syncthreads();

  // 2x2 maxpool (14x14 -> 7x7), store flat [16*49] into combined (K-padded row)
  for (int i = tid; i < 784; i += 128) {
    int ch = i / 49, q = i - 49 * ch, qh = q / 7, qw = q - 7 * qh;
    const float* yr = ys + ch * 208 + (2 * qh) * 14 + 2 * qw;
    float m = fmaxf(fmaxf(yr[0], yr[1]), fmaxf(yr[14], yr[15]));
    combined[(size_t)b * 800 + i] = (_Float16)m;
  }
}

// ---------------------------------------------------------------------------
// K3: per image (one wave32 each): sigmoid-mean of 2x2 conv over x (729 taps,
// cross-lane fixed reduction) + tanh/softmax sampler on combined[:,0:2].
// Writes combined cols 784..786, zeros 787..799.
// ---------------------------------------------------------------------------
__global__ void __launch_bounds__(256)
k3_sampler(const float* __restrict__ x, const float* __restrict__ s1w,
           const float* __restrict__ s1b, const float* __restrict__ s2w,
           const float* __restrict__ s2b, const float* __restrict__ fw,
           const float* __restrict__ fb, _Float16* __restrict__ combined) {
  int lane = threadIdx.x & 31;
  int b = blockIdx.x * 8 + (threadIdx.x >> 5);
  const float* xb = x + (size_t)b * 784;
  float w00 = fw[0], w01 = fw[1], w10 = fw[2], w11 = fw[3], bb = fb[0];
  float s = 0.0f;
  for (int i = lane; i < 729; i += 32) {
    int oh = i / 27, ow = i - 27 * oh;
    const float* xp = xb + oh * 28 + ow;
    float v = w00 * xp[0] + w01 * xp[1] + w10 * xp[28] + w11 * xp[29] + bb;
    s += 1.0f / (1.0f + expf(-v));
  }
#pragma unroll
  for (int off = 16; off > 0; off >>= 1) s += __shfl_xor(s, off, 32);
  if (lane == 0) {
    float cmean = s / 729.0f;
    _Float16* cb = combined + (size_t)b * 800;
    float f0 = (float)cb[0], f1 = (float)cb[1];
    float h[4];
#pragma unroll
    for (int i = 0; i < 4; ++i)
      h[i] = tanhf(f0 * s1w[i * 2 + 0] + f1 * s1w[i * 2 + 1] + s1b[i]);
    float l0 = s2b[0], l1 = s2b[1];
#pragma unroll
    for (int i = 0; i < 4; ++i) { l0 += h[i] * s2w[i]; l1 += h[i] * s2w[4 + i]; }
    float mx = fmaxf(l0, l1);
    float e0 = expf(l0 - mx), e1 = expf(l1 - mx), inv = 1.0f / (e0 + e1);
    cb[784] = (_Float16)(e0 * inv);
    cb[785] = (_Float16)(e1 * inv);
    cb[786] = (_Float16)cmean;
#pragma unroll
    for (int k = 787; k < 800; ++k) cb[k] = (_Float16)0.0f;
  }
}

// ---------------------------------------------------------------------------
// K4: fc1 (WMMA GEMM [B,800]x[800,64]) + relu + fc2 (VALU, 64->4) + per-block
// deterministic BN partial sums.  Block = 128 threads = 4 waves, 64 rows.
// ---------------------------------------------------------------------------
__global__ void __launch_bounds__(128)
k4_fc(const _Float16* __restrict__ combined, const _Float16* __restrict__ w1h,
      const float* __restrict__ fc1b, const float* __restrict__ fc2w,
      const float* __restrict__ fc2b, float* __restrict__ out_ws,
      float* __restrict__ partials) {
  __shared__ float ld1[4096];                        // 4 waves x 16 x 64 relu(fc1)
  __shared__ float outs[256];                        // 64 rows x 4 outputs
  int tid = threadIdx.x, lane = tid & 31, wv = tid >> 5;
  int l15 = lane & 15, hi = lane >> 4;
  int rowbase = blockIdx.x * 64 + wv * 16;
  int brow = rowbase + l15;                          // A: M = batch row
  const _Float16* arow = combined + (size_t)brow * 800 + hi * 16;

  v8f z = {0.f, 0.f, 0.f, 0.f, 0.f, 0.f, 0.f, 0.f};
  v8f acc[4] = {z, z, z, z};
  for (int s = 0; s < 25; ++s) {
    HU a;                                            // A: K = hi*16+e
    const h8* ap = (const h8*)(arow + s * 32);
    a.h[0] = ap[0];                                  // k = s*32+hi*16   .. +7
    a.h[1] = ap[1];                                  // k = s*32+hi*16+8 .. +15
#pragma unroll
    for (int t = 0; t < 4; ++t) {
      HU bb;                                         // B: pre-swizzled contiguous
      const h8* bp =
          (const h8*)(w1h + ((size_t)(s * 2 + hi) * 64 + t * 16 + l15) * 16);
      bb.h[0] = bp[0];
      bb.h[1] = bp[1];
      acc[t] = __builtin_amdgcn_wmma_f32_16x16x32_f16(false, a.v, false, bb.v,
                                                      (short)0, acc[t], false,
                                                      false);
    }
  }
  // bias + relu, scatter to LDS in [m][n] order
#pragma unroll
  for (int t = 0; t < 4; ++t) {
    int n = t * 16 + l15;
    float bias = fc1b[n];
#pragma unroll
    for (int v = 0; v < 8; ++v)
      ld1[wv * 1024 + ((hi << 3) + v) * 64 + n] = fmaxf(acc[t][v] + bias, 0.0f);
  }
  __syncthreads();

  // fc2: each lane computes 2 of the wave's 16x4 outputs
#pragma unroll
  for (int u = 0; u < 2; ++u) {
    int idx = u * 32 + lane;                         // 0..63 = m*4+j
    int m = idx >> 2, j = idx & 3;
    const float* row = ld1 + wv * 1024 + m * 64;
    float val = fc2b[j];
#pragma unroll
    for (int n = 0; n < 64; ++n) val += row[n] * fc2w[j * 64 + n];
    out_ws[(size_t)(rowbase + m) * 4 + j] = val;
    outs[wv * 64 + idx] = val;
  }
  __syncthreads();

  // deterministic per-block BN partials (sum, sumsq) over the 64 rows
  if (tid < 4) {
    float s = 0.0f, q = 0.0f;
    for (int i = 0; i < 64; ++i) {
      float v = outs[(i >> 4) * 64 + (i & 15) * 4 + tid];
      s += v;
      q += v * v;
    }
    partials[blockIdx.x * 8 + tid] = s;
    partials[blockIdx.x * 8 + 4 + tid] = q;
  }
}

// ---------------------------------------------------------------------------
// K6: reduce 256 block partials -> scale/shift per output channel (biased var)
// ---------------------------------------------------------------------------
__global__ void k6_bnstats(const float* __restrict__ partials,
                           const float* __restrict__ g,
                           const float* __restrict__ bta,
                           float* __restrict__ scsh) {
  int j = threadIdx.x;
  if (j < 4) {
    float s = 0.0f, q = 0.0f;
    for (int i = 0; i < 256; ++i) {
      s += partials[i * 8 + j];
      q += partials[i * 8 + 4 + j];
    }
    float mu = s / (float)BATCH;
    float var = q / (float)BATCH - mu * mu;
    float sc = rsqrtf(var + 1e-5f) * g[j];
    scsh[j] = sc;
    scsh[4 + j] = bta[j] - mu * sc;
  }
}

// ---------------------------------------------------------------------------
// K7: apply batchnorm -> d_out (f32)
// ---------------------------------------------------------------------------
__global__ void k7_apply(const float* __restrict__ out_ws,
                         const float* __restrict__ scsh,
                         float* __restrict__ out) {
  int i = blockIdx.x * 256 + threadIdx.x;
  if (i < BATCH * 4) {
    int j = i & 3;
    out[i] = out_ws[i] * scsh[j] + scsh[4 + j];
  }
}

// ---------------------------------------------------------------------------
extern "C" void kernel_launch(void* const* d_in, const int* in_sizes, int n_in,
                              void* d_out, int out_size, void* d_ws,
                              size_t ws_size, hipStream_t stream) {
  const float* x       = (const float*)d_in[0];
  const float* conv1_w = (const float*)d_in[1];
  const float* conv1_b = (const float*)d_in[2];
  const float* conv2_w = (const float*)d_in[3];
  const float* conv2_b = (const float*)d_in[4];
  const float* s1_w    = (const float*)d_in[5];
  const float* s1_b    = (const float*)d_in[6];
  const float* s2_w    = (const float*)d_in[7];
  const float* s2_b    = (const float*)d_in[8];
  const float* filt_w  = (const float*)d_in[9];
  const float* filt_b  = (const float*)d_in[10];
  const float* fc1_w   = (const float*)d_in[11];
  const float* fc1_b   = (const float*)d_in[12];
  const float* fc2_w   = (const float*)d_in[13];
  const float* fc2_b   = (const float*)d_in[14];
  const float* bn_g    = (const float*)d_in[15];
  const float* bn_b    = (const float*)d_in[16];

  char* ws = (char*)d_ws;
  _Float16* f1p      = (_Float16*)(ws);                       // 64 MB
  _Float16* combined = (_Float16*)(ws + 67108864);            // 25 MB
  _Float16* w1h      = (_Float16*)(ws + 93323264);            // 100 KB
  float*    out_ws   = (float*)(ws + 93425664);               // 256 KB
  float*    partials = (float*)(ws + 93687808);               // 8 KB
  float*    scsh     = (float*)(ws + 93696000);               // 32 B
  _Float16* w2p      = (_Float16*)(ws + 93696064);            // 3 KB

  k0_prep_fc1<<<200, 256, 0, stream>>>(fc1_w, w1h);
  k0b_prep_conv2<<<6, 256, 0, stream>>>(conv2_w, w2p);
  k1_conv1<<<BATCH, 256, 0, stream>>>(x, conv1_w, conv1_b, f1p);
  k2_conv2<<<BATCH, 128, 0, stream>>>(f1p, w2p, conv2_b, combined);
  k3_sampler<<<BATCH / 8, 256, 0, stream>>>(x, s1_w, s1_b, s2_w, s2_b, filt_w,
                                            filt_b, combined);
  k4_fc<<<BATCH / 64, 128, 0, stream>>>(combined, w1h, fc1_b, fc2_w, fc2_b,
                                        out_ws, partials);
  k6_bnstats<<<1, 32, 0, stream>>>(partials, bn_g, bn_b, scsh);
  k7_apply<<<(BATCH * 4 + 255) / 256, 256, 0, stream>>>(out_ws, scsh,
                                                        (float*)d_out);
}
